// AttentionPool_75952201662547
// MI455X (gfx1250) — matrix operand order repaired and checked by reading, not compiled, not run on hardware
//
#include <hip/hip_runtime.h>
#include <hip/hip_bf16.h>

typedef __attribute__((ext_vector_type(16))) __bf16 v16bf;
typedef __attribute__((ext_vector_type(8)))  float  v8f;

#define DIM   256
#define HID   128
#define NSEG  1024

// ---- monotone float<->uint encoding for atomicMax on floats ----
__device__ __forceinline__ unsigned encf(float f) {
    unsigned u = __float_as_uint(f);
    return (u & 0x80000000u) ? ~u : (u | 0x80000000u);
}
__device__ __forceinline__ float decf(unsigned u) {
    return __uint_as_float((u & 0x80000000u) ? (u ^ 0x80000000u) : ~u);
}

// ---------------------------------------------------------------------------
// Kernel 0: repack W1 (256x128 f32, row-major) into bf16 in WMMA B-fragment
// order so each lane's 16-element fragment is 32 contiguous bytes.
// idx = ((nt*8 + kk)*32 + lane)*16 + j ;  K = kk*32 + (j>>3)*16 + (lane>>4)*8 + (j&7),
// N = nt*16 + (lane&15)   (ISA 7.12.2, 16-bit B 32x16 layout)
// ---------------------------------------------------------------------------
__global__ void prep_w1(const float* __restrict__ W1, __bf16* __restrict__ w1s) {
    int idx  = blockIdx.x * blockDim.x + threadIdx.x;   // 0..32767
    int j    = idx & 15;
    int lane = (idx >> 4) & 31;
    int kk   = (idx >> 9) & 7;
    int nt   = idx >> 12;
    int k    = kk * 32 + (j >> 3) * 16 + (lane >> 4) * 8 + (j & 7);
    int n    = nt * 16 + (lane & 15);
    w1s[idx] = (__bf16)W1[k * HID + n];
}

// ---------------------------------------------------------------------------
// Kernel 1: per-wave 16-row tile: logit = relu(x@W1 + b1) @ W2 + b2
// W1 fragments staged once per block into LDS (64 KB); B frags fed from
// ds_load_b128. 64x v_wmma_f32_16x16x32_bf16 per tile; relu/dot fused on
// D fragments. Global max via wave-shuffle + 64-slot scattered atomicMax.
// ---------------------------------------------------------------------------
__global__ __launch_bounds__(256) void mlp_logit(
    const float* __restrict__ x, const __bf16* __restrict__ w1s,
    const float* __restrict__ b1, const float* __restrict__ W2,
    const float* __restrict__ b2, float* __restrict__ wraw,
    unsigned* __restrict__ maxslots, int n)
{
    __shared__ __bf16 sW1[DIM * HID];   // 32768 bf16 = 64 KB
    {
        const uint4* src = (const uint4*)w1s;
        uint4*       dst = (uint4*)sW1;
#pragma unroll
        for (int i = 0; i < 16; ++i)    // 4096 uint4 total / 256 threads
            dst[threadIdx.x + 256 * i] = src[threadIdx.x + 256 * i];
    }
    __syncthreads();

    const int lane = threadIdx.x & 31;
    const int wave = threadIdx.x >> 5;
    const int tile = blockIdx.x * 8 + wave;
    const int row0 = tile * 16;
    const int np   = lane & 15;            // M for A, N for B/D lane mapping
    const int kh   = (lane >> 4) * 8;      // K sub-offset for A fragments

    int rowm = row0 + np;                  // clamp instead of masking (EXEC all-1s)
    if (rowm >= n) rowm = n - 1;
    const float* xr = x + (size_t)rowm * DIM;

    // A fragments: 16x32 bf16 per K-step, built from coalesced float4 loads
    v16bf a[8];
#pragma unroll
    for (int kk = 0; kk < 8; ++kk) {
        const float4 c0 = *reinterpret_cast<const float4*>(xr + kk * 32 + kh);
        const float4 c1 = *reinterpret_cast<const float4*>(xr + kk * 32 + kh + 4);
        const float4 c2 = *reinterpret_cast<const float4*>(xr + kk * 32 + 16 + kh);
        const float4 c3 = *reinterpret_cast<const float4*>(xr + kk * 32 + 16 + kh + 4);
        a[kk][0]  = (__bf16)c0.x; a[kk][1]  = (__bf16)c0.y;
        a[kk][2]  = (__bf16)c0.z; a[kk][3]  = (__bf16)c0.w;
        a[kk][4]  = (__bf16)c1.x; a[kk][5]  = (__bf16)c1.y;
        a[kk][6]  = (__bf16)c1.z; a[kk][7]  = (__bf16)c1.w;
        a[kk][8]  = (__bf16)c2.x; a[kk][9]  = (__bf16)c2.y;
        a[kk][10] = (__bf16)c2.z; a[kk][11] = (__bf16)c2.w;
        a[kk][12] = (__bf16)c3.x; a[kk][13] = (__bf16)c3.y;
        a[kk][14] = (__bf16)c3.z; a[kk][15] = (__bf16)c3.w;
    }

    float partial[8] = {0.f,0.f,0.f,0.f,0.f,0.f,0.f,0.f};
#pragma unroll
    for (int nt = 0; nt < 8; ++nt) {
        // hoist all 8 B fragments from LDS so the DS loads clause together
        v16bf bfr[8];
#pragma unroll
        for (int kk = 0; kk < 8; ++kk)
            bfr[kk] = *reinterpret_cast<const v16bf*>(
                sW1 + ((nt * 8 + kk) * 32 + lane) * 16);

        v8f acc = {0.f,0.f,0.f,0.f,0.f,0.f,0.f,0.f};
#pragma unroll
        for (int kk = 0; kk < 8; ++kk)
            acc = __builtin_amdgcn_wmma_f32_16x16x32_bf16(
                false, a[kk], false, bfr[kk], (short)0, acc, false, false);

        const int col   = nt * 16 + np;
        const float b1v = b1[col];
        const float w2v = W2[col];
#pragma unroll
        for (int r = 0; r < 8; ++r) {
            float hv = acc[r] + b1v;
            hv = hv > 0.f ? hv : 0.f;          // relu
            partial[r] += hv * w2v;            // h @ W2, N striped over lanes
        }
    }

    // reduce the 16 N-lanes within each half-wave
#pragma unroll
    for (int r = 0; r < 8; ++r) {
        float v = partial[r];
        v += __shfl_xor(v, 1);
        v += __shfl_xor(v, 2);
        v += __shfl_xor(v, 4);
        v += __shfl_xor(v, 8);
        partial[r] = v;
    }

    const float b2v = b2[0];
    float lmax = -__builtin_inff();
#pragma unroll
    for (int r = 0; r < 8; ++r) {
        partial[r] += b2v;
        lmax = fmaxf(lmax, partial[r]);
    }

    if (np == 0) {
        int mbase = row0 + (lane >> 4) * 8;    // rows 0..7 (lane0) / 8..15 (lane16)
#pragma unroll
        for (int r = 0; r < 8; ++r) {
            int row = mbase + r;
            if (row < n) wraw[row] = partial[r];
        }
    }

    // full-wave max, then one scattered atomic per wave (64 slots)
    lmax = fmaxf(lmax, __shfl_xor(lmax, 16));
    if (lane == 0 && row0 < n)
        atomicMax(&maxslots[blockIdx.x & 63], encf(lmax));
}

// ---------------------------------------------------------------------------
// Kernel 1b: collapse the 64 partial-max slots into maxslots[64]
// ---------------------------------------------------------------------------
__global__ void max_reduce(unsigned* __restrict__ maxslots) {
    unsigned v = maxslots[threadIdx.x];                    // 32 threads
    unsigned w = maxslots[threadIdx.x + 32];
    v = v > w ? v : w;
#pragma unroll
    for (int off = 16; off > 0; off >>= 1) {
        unsigned o = (unsigned)__shfl_xor((int)v, off);
        v = v > o ? v : o;
    }
    if (threadIdx.x == 0) maxslots[64] = v;
}

// ---------------------------------------------------------------------------
// Kernel 2: w = exp(logit - max); per-segment sum & count (f32 atomics)
// ---------------------------------------------------------------------------
__global__ void seg_exp_sum(const float* __restrict__ wraw, const int* __restrict__ batch,
                            const unsigned* __restrict__ maxenc, float* __restrict__ wbuf,
                            float* __restrict__ segsum, float* __restrict__ segcnt, int n)
{
    int i = blockIdx.x * blockDim.x + threadIdx.x;
    if (i >= n) return;
    float mx = decf(*maxenc);
    float w  = __expf(wraw[i] - mx);
    wbuf[i]  = w;
    int s = batch[i];
    atomicAdd(&segsum[s], w);
    atomicAdd(&segcnt[s], 1.0f);
}

// ---------------------------------------------------------------------------
// Kernel 2b: coef = w / (mean_w[seg] * N + 1e-8)
// ---------------------------------------------------------------------------
__global__ void coef_k(float* __restrict__ wbuf, const int* __restrict__ batch,
                       const float* __restrict__ segsum, const float* __restrict__ segcnt,
                       int n)
{
    int i = blockIdx.x * blockDim.x + threadIdx.x;
    if (i >= n) return;
    int s = batch[i];
    float mean  = segsum[s] / fmaxf(segcnt[s], 1.0f);
    float denom = mean * (float)n;
    wbuf[i] = wbuf[i] / (denom + 1e-8f);
}

// ---------------------------------------------------------------------------
// Kernel 3: pooled numerator: 256 rows per block, thread = column; sorted
// batch => register accumulation with atomic flush only on segment change.
// ---------------------------------------------------------------------------
__global__ __launch_bounds__(256) void pool_acc(
    const float* __restrict__ x, const int* __restrict__ batch,
    const float* __restrict__ coef, float* __restrict__ pool, int n)
{
    __shared__ int   sseg[256];
    __shared__ float scoef[256];
    const int r0 = blockIdx.x * 256;
    const int t  = threadIdx.x;
    int nr = n - r0; if (nr > 256) nr = 256;
    if (t < nr) { sseg[t] = batch[r0 + t]; scoef[t] = coef[r0 + t]; }
    __syncthreads();

    float acc = 0.f;
    int cur = sseg[0];
    for (int i = 0; i < nr; ++i) {
        int s = sseg[i];
        if (s != cur) {
            atomicAdd(&pool[(size_t)cur * DIM + t], acc);
            acc = 0.f;
            cur = s;
        }
        acc += scoef[i] * x[(size_t)(r0 + i) * DIM + t];
    }
    atomicAdd(&pool[(size_t)cur * DIM + t], acc);
}

// ---------------------------------------------------------------------------
// Kernel 4: pooled = pool_sum / max(count, 1)
// ---------------------------------------------------------------------------
__global__ void finalize(const float* __restrict__ pool, const float* __restrict__ segcnt,
                         float* __restrict__ out)
{
    int i = blockIdx.x * blockDim.x + threadIdx.x;   // 0 .. NSEG*DIM-1
    int b = i >> 8;                                  // DIM == 256
    out[i] = pool[i] / fmaxf(segcnt[b], 1.0f);
}

// ---------------------------------------------------------------------------
extern "C" void kernel_launch(void* const* d_in, const int* in_sizes, int n_in,
                              void* d_out, int out_size, void* d_ws, size_t ws_size,
                              hipStream_t stream)
{
    const float* x     = (const float*)d_in[0];
    const int*   batch = (const int*)d_in[1];
    const float* W1    = (const float*)d_in[2];
    const float* b1    = (const float*)d_in[3];
    const float* W2    = (const float*)d_in[4];
    const float* b2    = (const float*)d_in[5];
    float* out = (float*)d_out;
    const int n = in_sizes[0] / DIM;   // N = 500000

    // workspace carve-up (all offsets 256B-aligned)
    char* ws = (char*)d_ws;
    float* wbuf = (float*)ws;                                 // N floats (logit->w->coef)
    size_t zoff = ((size_t)n * 4 + 255) & ~(size_t)255;
    float*    segsum   = (float*)(ws + zoff);                 // NSEG floats
    float*    segcnt   = (float*)(ws + zoff + 4096);          // NSEG floats
    unsigned* maxslots = (unsigned*)(ws + zoff + 8192);       // 64 partial + [64] final (512B)
    float*    pool     = (float*)(ws + zoff + 8704);          // NSEG*DIM floats
    __bf16*   w1s      = (__bf16*)(ws + zoff + 8704 + (size_t)NSEG * DIM * 4);

    // zero segsum/segcnt/maxslots/pool in one shot (enc==0 is the -max floor)
    hipMemsetAsync(ws + zoff, 0, 8704 + (size_t)NSEG * DIM * 4, stream);

    prep_w1<<<(DIM * HID) / 256, 256, 0, stream>>>(W1, w1s);

    const int tiles = (n + 15) / 16;
    mlp_logit<<<(tiles + 7) / 8, 256, 0, stream>>>(x, w1s, b1, W2, b2, wbuf, maxslots, n);
    max_reduce<<<1, 32, 0, stream>>>(maxslots);

    const int blk = (n + 255) / 256;
    seg_exp_sum<<<blk, 256, 0, stream>>>(wbuf, batch, &maxslots[64], wbuf, segsum, segcnt, n);
    coef_k<<<blk, 256, 0, stream>>>(wbuf, batch, segsum, segcnt, n);
    pool_acc<<<blk, 256, 0, stream>>>(x, batch, wbuf, pool, n);
    finalize<<<(NSEG * DIM) / 256, 256, 0, stream>>>(pool, segcnt, out);
}
